// BertBiAttention_76630806495508
// MI455X (gfx1250) — matrix operand order repaired and checked by reading, not compiled, not run on hardware
//
#include <hip/hip_runtime.h>
#include <hip/hip_bf16.h>

typedef _Float16 h16;
typedef _Float16 half8  __attribute__((ext_vector_type(8)));
typedef _Float16 half16 __attribute__((ext_vector_type(16)));
typedef float    f32x8  __attribute__((ext_vector_type(8)));

#define B_    4
#define S_    2048
#define HID_  768
#define H_    12
#define HD_   64
#define MTOT  (B_ * S_)        /* 8192 */
#define NELEM (MTOT * HID_)    /* 6291456 */
#define LOG2E 1.44269504088896340736f

static __device__ __forceinline__ half16 mk16(half8 lo, half8 hi) {
  half16 r;
#pragma unroll
  for (int i = 0; i < 8; i++) { r[i] = lo[i]; r[i + 8] = hi[i]; }
  return r;
}

static __device__ __forceinline__ f32x8 wmma_f16(half16 a, half16 b, f32x8 c) {
  return __builtin_amdgcn_wmma_f32_16x16x32_f16(false, a, false, b,
                                                (short)0, c, false, false);
}

// Async 16B copy global -> LDS (CDNA5 GLOBAL_LOAD_ASYNC_TO_LDS_B128, ASYNCcnt).
static __device__ __forceinline__ void async_copy_b128(void* lds_dst,
                                                       const void* gsrc) {
  unsigned laddr = (unsigned)(unsigned long long)(uintptr_t)lds_dst;
  asm volatile("global_load_async_to_lds_b128 %0, %1, off"
               :: "v"(laddr), "v"(gsrc)
               : "memory");
}
static __device__ __forceinline__ void wait_async0() {
  asm volatile("s_wait_asynccnt 0x0" ::: "memory");
}

// ---- DPP16 16-lane butterfly reductions (stay inside each 16-lane DPP row,
// which matches our row-group split: lanes 0-15 vs 16-31 hold different M). ----
template <int CTRL>
static __device__ __forceinline__ float dpp_xchg(float x) {
  return __int_as_float(__builtin_amdgcn_update_dpp(
      __float_as_int(x), __float_as_int(x), CTRL, 0xF, 0xF, false));
}
static __device__ __forceinline__ float red_max16(float x) {
  x = fmaxf(x, dpp_xchg<0xB1>(x));    // quad_perm [1,0,3,2] : xor1
  x = fmaxf(x, dpp_xchg<0x4E>(x));    // quad_perm [2,3,0,1] : xor2
  x = fmaxf(x, dpp_xchg<0x141>(x));   // ROW_HALF_MIRROR     : 8-group mix
  x = fmaxf(x, dpp_xchg<0x140>(x));   // ROW_MIRROR          : 16-group mix
  return x;
}
static __device__ __forceinline__ float red_sum16(float x) {
  x += dpp_xchg<0xB1>(x);
  x += dpp_xchg<0x4E>(x);
  x += dpp_xchg<0x141>(x);
  x += dpp_xchg<0x140>(x);
  return x;
}

// ---------------- fp32 -> f16 conversion ----------------
__global__ void cvt_h_kernel(const float* __restrict__ in,
                             h16* __restrict__ out, int n) {
  int i = blockIdx.x * blockDim.x + threadIdx.x;
  int stride = gridDim.x * blockDim.x;
  for (; i < n; i += stride) out[i] = (h16)in[i];
}

// transpose + convert weight [K=768][N=768] -> Wt f16 [N][K]
__global__ void cvt_wt_kernel(const float* __restrict__ w,
                              h16* __restrict__ wt) {
  int i = blockIdx.x * blockDim.x + threadIdx.x;  // exactly 768*768 threads
  int n = i / HID_;
  int k = i - n * HID_;
  wt[i] = (h16)w[k * HID_ + n];
}

// ---------------- WMMA GEMM: C[M=8192, N=768] = X f16 * Wt^T + bias --------
// MODE 0: store f16 head layout   q[((b*H+h)*S + s)*HD + d]
// MODE 1: store f16 head-T layout v[((b*H+h)*HD + d)*S + s]
// MODE 2: store f32 flat          y[m*HID + n]
template <int MODE>
__global__ __launch_bounds__(256) void gemm_wmma_kernel(
    const h16* __restrict__ X,    // [M][768]
    const h16* __restrict__ Wt,   // [N=768][K=768]  (W transposed)
    const float* __restrict__ bias,
    void* __restrict__ outp) {
  __shared__ h16 As[2][64][40];    // pitch 80B: 16B aligned, bank-spread
  __shared__ h16 Bs[2][128][40];

  const int tid  = threadIdx.x;
  const int lane = tid & 31;
  const int w    = tid >> 5;       // 8 waves
  const int l    = lane & 15;
  const int hi   = lane >> 4;
  const int mBase = blockIdx.y * 64;
  const int nBase = blockIdx.x * 128;
  const int wm = w >> 2;           // 0..1
  const int wn = w & 3;            // 0..3
  const int rowA = tid >> 2, partA = tid & 3;

  auto stage = [&](int buf, int k0) {
    async_copy_b128(&As[buf][rowA][partA * 8],
                    X + (size_t)(mBase + rowA) * HID_ + k0 + partA * 8);
#pragma unroll
    for (int i = 0; i < 2; i++) {
      int ch = tid + i * 256;
      int row = ch >> 2, part = ch & 3;
      async_copy_b128(&Bs[buf][row][part * 8],
                      Wt + (size_t)(nBase + row) * HID_ + k0 + part * 8);
    }
  };

  f32x8 zero = {};
  f32x8 acc[2][2];
#pragma unroll
  for (int i = 0; i < 2; i++)
#pragma unroll
    for (int j = 0; j < 2; j++) acc[i][j] = zero;

  stage(0, 0);
  int buf = 0;
  for (int k0 = 0; k0 < HID_; k0 += 32, buf ^= 1) {
    wait_async0();
    __syncthreads();
    if (k0 + 32 < HID_) stage(buf ^ 1, k0 + 32);  // prefetch next tile

    half16 a[2], bf[2];
#pragma unroll
    for (int i = 0; i < 2; i++) {   // A frag: lane=M, K 0-7/16-23 (hi=0)
      int mr = wm * 32 + i * 16 + l;
      a[i] = mk16(*(const half8*)&As[buf][mr][hi * 8],
                  *(const half8*)&As[buf][mr][16 + hi * 8]);
    }
#pragma unroll
    for (int j = 0; j < 2; j++) {   // B frag: lane=N, K 0-7/16-23 (hi=0)
      int nr = wn * 32 + j * 16 + l;
      bf[j] = mk16(*(const half8*)&Bs[buf][nr][hi * 8],
                   *(const half8*)&Bs[buf][nr][16 + hi * 8]);
    }
#pragma unroll
    for (int i = 0; i < 2; i++)
#pragma unroll
      for (int j = 0; j < 2; j++) acc[i][j] = wmma_f16(a[i], bf[j], acc[i][j]);
  }

  // store (C layout: vgpr r -> M = hi*8 + r, lane -> N = l)
#pragma unroll
  for (int j = 0; j < 2; j++) {
    int n = nBase + wn * 32 + j * 16 + l;
    float bj = bias[n];
#pragma unroll
    for (int i = 0; i < 2; i++) {
      int mTile = mBase + wm * 32 + i * 16 + hi * 8;
#pragma unroll
      for (int r = 0; r < 8; r++) {
        int m = mTile + r;
        float v = acc[i][j][r] + bj;
        if constexpr (MODE == 2) {
          ((float*)outp)[(size_t)m * HID_ + n] = v;
        } else {
          int b = m >> 11, s = m & 2047;
          int h = n >> 6, d = n & 63;
          h16* o = (h16*)outp;
          if constexpr (MODE == 0)
            o[(((size_t)(b * H_ + h)) * S_ + s) * HD_ + d] = (h16)v;
          else
            o[(((size_t)(b * H_ + h)) * HD_ + d) * S_ + s] = (h16)v;
        }
      }
    }
  }
}

// ---------------- flash attention: ctx = softmax(Q K^T / 8 + mask) V -------
__global__ __launch_bounds__(128) void attn_wmma_kernel(
    const h16* __restrict__ Q,    // [BH][S][64]
    const h16* __restrict__ K,    // [BH][S][64]
    const h16* __restrict__ VT,   // [BH][64][S]
    const float* __restrict__ mask,  // [B][S] additive
    h16* __restrict__ ctx) {      // [B][S][768]
  __shared__ h16 Qs[64][72];           // pitch 144B (16B aligned)
  __shared__ h16 Ks[2][32][72];
  __shared__ h16 Vs[2][64][40];
  __shared__ h16 Ps[4][16][40];        // per-wave probs scratch

  const int tid  = threadIdx.x;
  const int lane = tid & 31;
  const int w    = tid >> 5;   // 4 waves, 16 q rows each
  const int l    = lane & 15;
  const int hi   = lane >> 4;
  const int bh   = blockIdx.y;
  const int b    = bh / H_;
  const int h    = bh - b * H_;
  const int qBase = blockIdx.x * 64;
  const size_t qkOff = (size_t)bh * S_ * HD_;

  auto stageKV = [&](int buf, int kb) {
#pragma unroll
    for (int i = 0; i < 2; i++) {  // K tile 32x64
      int ch = tid + i * 128;
      int row = ch >> 3, part = ch & 7;
      async_copy_b128(&Ks[buf][row][part * 8],
                      K + qkOff + (size_t)(kb + row) * HD_ + part * 8);
    }
#pragma unroll
    for (int i = 0; i < 2; i++) {  // V^T tile 64x32
      int ch = tid + i * 128;
      int row = ch >> 2, part = ch & 3;
      async_copy_b128(&Vs[buf][row][part * 8],
                      VT + qkOff + (size_t)row * S_ + kb + part * 8);
    }
  };

#pragma unroll
  for (int i = 0; i < 4; i++) {  // stage 64x64 Q tile (async)
    int ch = tid + i * 128;
    int row = ch >> 3, part = ch & 7;
    async_copy_b128(&Qs[row][part * 8],
                    Q + qkOff + (size_t)(qBase + row) * HD_ + part * 8);
  }
  stageKV(0, 0);
  wait_async0();
  __syncthreads();

  half16 aq0, aq1;
  {
    int mr = w * 16 + l;
    aq0 = mk16(*(const half8*)&Qs[mr][hi * 8],
               *(const half8*)&Qs[mr][16 + hi * 8]);
    aq1 = mk16(*(const half8*)&Qs[mr][32 + hi * 8],
               *(const half8*)&Qs[mr][48 + hi * 8]);
  }

  f32x8 zero = {};
  f32x8 O[4];
#pragma unroll
  for (int t = 0; t < 4; t++) O[t] = zero;
  float mrun[8], lrun[8];
#pragma unroll
  for (int r = 0; r < 8; r++) { mrun[r] = -1e30f; lrun[r] = 0.f; }
  const float scale2 = 0.125f * LOG2E;  // scores in log2 units

  int buf = 0;
  for (int kb = 0; kb < S_; kb += 32, buf ^= 1) {
    wait_async0();
    __syncthreads();
    if (kb + 32 < S_) stageKV(buf ^ 1, kb + 32);  // prefetch next key block

    // load all 4 B frags, then issue the 4 score WMMAs together
    half16 bk[2][2];
#pragma unroll
    for (int nt = 0; nt < 2; nt++) {
      int nr = nt * 16 + l;
      bk[nt][0] = mk16(*(const half8*)&Ks[buf][nr][hi * 8],
                       *(const half8*)&Ks[buf][nr][16 + hi * 8]);
      bk[nt][1] = mk16(*(const half8*)&Ks[buf][nr][32 + hi * 8],
                       *(const half8*)&Ks[buf][nr][48 + hi * 8]);
    }
    f32x8 s[2];
#pragma unroll
    for (int nt = 0; nt < 2; nt++) {
      f32x8 t = zero;
      t = wmma_f16(aq0, bk[nt][0], t);
      t = wmma_f16(aq1, bk[nt][1], t);
      s[nt] = t;
    }
    float mk0 = mask[b * S_ + kb + l] * LOG2E;
    float mk1 = mask[b * S_ + kb + 16 + l] * LOG2E;

#pragma unroll
    for (int r = 0; r < 8; r++) {  // online softmax per row (base-2)
      float t0 = s[0][r] * scale2 + mk0;
      float t1 = s[1][r] * scale2 + mk1;
      float mx = red_max16(fmaxf(t0, t1));
      float mnew = fmaxf(mrun[r], mx);
      float corr = __builtin_amdgcn_exp2f(mrun[r] - mnew);
      float p0 = __builtin_amdgcn_exp2f(t0 - mnew);
      float p1 = __builtin_amdgcn_exp2f(t1 - mnew);
      float rs = red_sum16(p0 + p1);
      lrun[r] = lrun[r] * corr + rs;
      mrun[r] = mnew;
#pragma unroll
      for (int t = 0; t < 4; t++) O[t][r] *= corr;
      Ps[w][hi * 8 + r][l]      = (h16)p0;   // C-layout -> LDS
      Ps[w][hi * 8 + r][16 + l] = (h16)p1;
    }
    asm volatile("s_wait_dscnt 0x0" ::: "memory");

    // probs A-frag + all 4 V B-frags, then 4 ctx WMMAs together
    half16 ap = mk16(*(const half8*)&Ps[w][l][hi * 8],
                     *(const half8*)&Ps[w][l][16 + hi * 8]);
    half16 bv[4];
#pragma unroll
    for (int t = 0; t < 4; t++) {
      int nr = t * 16 + l;
      bv[t] = mk16(*(const half8*)&Vs[buf][nr][hi * 8],
                   *(const half8*)&Vs[buf][nr][16 + hi * 8]);
    }
#pragma unroll
    for (int t = 0; t < 4; t++) O[t] = wmma_f16(ap, bv[t], O[t]);
  }

#pragma unroll
  for (int r = 0; r < 8; r++) {
    float inv = 1.0f / lrun[r];
    int m = qBase + w * 16 + hi * 8 + r;
    size_t rowOff = ((size_t)b * S_ + m) * HID_ + h * HD_;
#pragma unroll
    for (int t = 0; t < 4; t++)
      ctx[rowOff + t * 16 + l] = (h16)(O[t][r] * inv);
  }
}

// ---------------- residual + LayerNorm ----------------
__global__ __launch_bounds__(256) void ln_kernel(
    const float* __restrict__ y, const float* __restrict__ x,
    const float* __restrict__ g, const float* __restrict__ bb,
    float* __restrict__ out) {
  __shared__ float red[256];
  const int row = blockIdx.x;
  const int t = threadIdx.x;
  const float* yr = y + (size_t)row * HID_;
  const float* xr = x + (size_t)row * HID_;
  float v[3];
  float s = 0.f, sq = 0.f;
#pragma unroll
  for (int i = 0; i < 3; i++) {
    int c = t + i * 256;
    float vv = yr[c] + xr[c];
    v[i] = vv; s += vv; sq += vv * vv;
  }
  red[t] = s; __syncthreads();
  for (int off = 128; off > 0; off >>= 1) {
    if (t < off) red[t] += red[t + off];
    __syncthreads();
  }
  float mu = red[0] * (1.0f / 768.0f);
  __syncthreads();
  red[t] = sq; __syncthreads();
  for (int off = 128; off > 0; off >>= 1) {
    if (t < off) red[t] += red[t + off];
    __syncthreads();
  }
  float var = red[0] * (1.0f / 768.0f) - mu * mu;
  float inv = rsqrtf(var + 1e-12f);
#pragma unroll
  for (int i = 0; i < 3; i++) {
    int c = t + i * 256;
    out[(size_t)row * HID_ + c] = (v[i] - mu) * inv * g[c] + bb[c];
  }
}

extern "C" void kernel_launch(void* const* d_in, const int* in_sizes, int n_in,
                              void* d_out, int out_size, void* d_ws,
                              size_t ws_size, hipStream_t stream) {
  (void)in_sizes; (void)n_in; (void)out_size; (void)ws_size;
  const float* x1   = (const float*)d_in[0];
  const float* msk1 = (const float*)d_in[1];
  const float* x2   = (const float*)d_in[2];
  const float* msk2 = (const float*)d_in[3];
  const float* wq1 = (const float*)d_in[4];  const float* bq1 = (const float*)d_in[5];
  const float* wk1 = (const float*)d_in[6];  const float* bk1 = (const float*)d_in[7];
  const float* wv1 = (const float*)d_in[8];  const float* bv1 = (const float*)d_in[9];
  const float* wq2 = (const float*)d_in[10]; const float* bq2 = (const float*)d_in[11];
  const float* wk2 = (const float*)d_in[12]; const float* bk2 = (const float*)d_in[13];
  const float* wv2 = (const float*)d_in[14]; const float* bv2 = (const float*)d_in[15];
  const float* wd1 = (const float*)d_in[16]; const float* bd1 = (const float*)d_in[17];
  const float* wd2 = (const float*)d_in[18]; const float* bd2 = (const float*)d_in[19];
  const float* g1  = (const float*)d_in[20]; const float* be1 = (const float*)d_in[21];
  const float* g2  = (const float*)d_in[22]; const float* be2 = (const float*)d_in[23];

  char* ws = (char*)d_ws;
  const size_t F16SZ = (size_t)NELEM * 2;          // 12,582,912 B
  const size_t WTSZ  = (size_t)HID_ * HID_ * 2;    // 1,179,648 B
  h16* x1h = (h16*)(ws);
  h16* x2h = (h16*)(ws + F16SZ);
  char* wtb = ws + 2 * F16SZ;
  h16* wtq1 = (h16*)(wtb + 0 * WTSZ);
  h16* wtk1 = (h16*)(wtb + 1 * WTSZ);
  h16* wtv1 = (h16*)(wtb + 2 * WTSZ);
  h16* wtq2 = (h16*)(wtb + 3 * WTSZ);
  h16* wtk2 = (h16*)(wtb + 4 * WTSZ);
  h16* wtv2 = (h16*)(wtb + 5 * WTSZ);
  h16* wtd1 = (h16*)(wtb + 6 * WTSZ);
  h16* wtd2 = (h16*)(wtb + 7 * WTSZ);
  char* qkvb = wtb + 8 * WTSZ;
  h16* q1  = (h16*)(qkvb + 0 * F16SZ);
  h16* k1  = (h16*)(qkvb + 1 * F16SZ);
  h16* v1T = (h16*)(qkvb + 2 * F16SZ);
  h16* q2  = (h16*)(qkvb + 3 * F16SZ);
  h16* k2  = (h16*)(qkvb + 4 * F16SZ);
  h16* v2T = (h16*)(qkvb + 5 * F16SZ);
  // alias reuse (sequential stream ordering makes this safe):
  h16* ctx1h = x1h;                        // x1h dead after projections
  h16* ctx2h = x2h;
  float* yd1 = (float*)(qkvb + 1 * F16SZ); // over k1+v1T, dead after attn1
  float* yd2 = (float*)(qkvb + 4 * F16SZ); // over k2+v2T, dead after attn2

  // 1) conversions
  cvt_h_kernel<<<4096, 256, 0, stream>>>(x1, x1h, NELEM);
  cvt_h_kernel<<<4096, 256, 0, stream>>>(x2, x2h, NELEM);
  const int WTG = (HID_ * HID_) / 256;  // 2304
  cvt_wt_kernel<<<WTG, 256, 0, stream>>>(wq1, wtq1);
  cvt_wt_kernel<<<WTG, 256, 0, stream>>>(wk1, wtk1);
  cvt_wt_kernel<<<WTG, 256, 0, stream>>>(wv1, wtv1);
  cvt_wt_kernel<<<WTG, 256, 0, stream>>>(wq2, wtq2);
  cvt_wt_kernel<<<WTG, 256, 0, stream>>>(wk2, wtk2);
  cvt_wt_kernel<<<WTG, 256, 0, stream>>>(wv2, wtv2);
  cvt_wt_kernel<<<WTG, 256, 0, stream>>>(wd1, wtd1);
  cvt_wt_kernel<<<WTG, 256, 0, stream>>>(wd2, wtd2);

  // 2) QKV projections
  dim3 gg(HID_ / 128, MTOT / 64);  // (6,128)
  gemm_wmma_kernel<0><<<gg, 256, 0, stream>>>(x1h, wtq1, bq1, q1);
  gemm_wmma_kernel<0><<<gg, 256, 0, stream>>>(x1h, wtk1, bk1, k1);
  gemm_wmma_kernel<1><<<gg, 256, 0, stream>>>(x1h, wtv1, bv1, v1T);
  gemm_wmma_kernel<0><<<gg, 256, 0, stream>>>(x2h, wtq2, bq2, q2);
  gemm_wmma_kernel<0><<<gg, 256, 0, stream>>>(x2h, wtk2, bk2, k2);
  gemm_wmma_kernel<1><<<gg, 256, 0, stream>>>(x2h, wtv2, bv2, v2T);

  // 3) cross attention: ctx1 = attend(q2, k1, v1); ctx2 = attend(q1, k2, v2)
  dim3 ga(S_ / 64, B_ * H_);  // (32,48)
  attn_wmma_kernel<<<ga, 128, 0, stream>>>(q2, k1, v1T, msk1, ctx1h);
  attn_wmma_kernel<<<ga, 128, 0, stream>>>(q1, k2, v2T, msk2, ctx2h);

  // 4) output projections (f32) then residual + LN into d_out
  gemm_wmma_kernel<2><<<gg, 256, 0, stream>>>(ctx1h, wtd1, bd1, yd1);
  gemm_wmma_kernel<2><<<gg, 256, 0, stream>>>(ctx2h, wtd2, bd2, yd2);
  ln_kernel<<<MTOT, 256, 0, stream>>>(yd1, x1, g1, be1, (float*)d_out);
  ln_kernel<<<MTOT, 256, 0, stream>>>(yd2, x2, g2, be2, (float*)d_out + NELEM);
}
